// SparseAttention_27908697489500
// MI455X (gfx1250) — matrix operand order repaired and checked by reading, not compiled, not run on hardware
//
#include <hip/hip_runtime.h>
#include <stdint.h>

#define BB   2
#define NN   2048
#define DIM  512
#define HH   8
#define HD   64
#define TOPK 32
#define MM   (BB * NN)            // 4096 rows for the big GEMMs
#define SCALE 0.125f              // 1/sqrt(64)

#define WMT  2                    // 16-row subtiles per wave  (32 rows)
#define WNT  4                    // 16-col subtiles per wave  (64 cols)

typedef __attribute__((ext_vector_type(16))) __bf16 v16bf;
typedef __attribute__((ext_vector_type(8)))  float  v8f;
typedef __attribute__((ext_vector_type(4)))  unsigned u32x4;
typedef unsigned short u16;

union Frag16 { v16bf v; u32x4 q[2]; };

__device__ inline float bf2f(u16 h) {
    union { unsigned u; float f; } c; c.u = (unsigned)h << 16; return c.f;
}
__device__ inline u16 f2bf(float f) {
    union { float f; unsigned u; } c; c.f = f;
    unsigned r = c.u + 0x7FFFu + ((c.u >> 16) & 1u);   // round-to-nearest-even
    return (u16)(r >> 16);
}

// A fragment: 16x32 bf16, row m = lane&15; lanes 0-15 hold K {0..7,16..23},
// lanes 16-31 hold K {8..15,24..31}  (ISA 7.12.2, 16-bit A 16x32)
__device__ inline v16bf load_frag_A(const u16* __restrict__ X, int ld, int kk, int lane) {
    int m  = lane & 15;
    int b0 = (lane < 16) ? 0 : 8;
    const u16* p = X + (size_t)m * ld + kk;
    Frag16 f;
    f.q[0] = *(const u32x4*)(p + b0);
    f.q[1] = *(const u32x4*)(p + b0 + 16);
    return f.v;
}
// B fragment: 32x16 bf16 (B = W^T, W row-major [n][k]); col n = lane&15,
// lanes 0-15 hold K {0..7,16..23}, lanes 16-31 hold K {8..15,24..31}
__device__ inline v16bf load_frag_B(const u16* __restrict__ W, int ld, int kk, int lane) {
    int n  = lane & 15;
    int kb = (lane < 16) ? 0 : 8;
    const u16* p = W + (size_t)n * ld + kk + kb;
    Frag16 f;
    f.q[0] = *(const u32x4*)(p);
    f.q[1] = *(const u32x4*)(p + 16);
    return f.v;
}

__global__ void f32_to_bf16_kernel(const float* __restrict__ src, u16* __restrict__ dst, int n) {
    int i = blockIdx.x * blockDim.x + threadIdx.x;
    if (i < n) dst[i] = f2bf(src[i]);
}

// O = A(bf16,[M,Kd]) * W(bf16,[Nout,Kd])^T + bias, register-blocked 2x4
// (each wave computes a 32x64 output tile: 8 WMMAs per 32-wide K step).
// mode 0: O fp32 row-major [M,Nout]
// mode 1: O bf16 split-head layout [B,H,N,HD]  (M = B*N, Nout = H*HD)
__global__ void wmma_gemm_kernel(const u16* __restrict__ A, const u16* __restrict__ W,
                                 const float* __restrict__ bias, void* __restrict__ out,
                                 int M, int Kd, int Nout, int mode)
{
    int nsuper = Nout / (16 * WNT);
    int tile   = blockIdx.x * (blockDim.x >> 5) + (threadIdx.x >> 5);
    int lane   = threadIdx.x & 31;
    int mt = tile / nsuper;
    int nt = tile % nsuper;
    if (mt >= M / (16 * WMT)) return;       // wave-uniform

    const u16* Arow = A + (size_t)(mt * 16 * WMT) * Kd;
    const u16* Wrow = W + (size_t)(nt * 16 * WNT) * Kd;

    v8f acc[WMT][WNT] = {};
    for (int kk = 0; kk < Kd; kk += 32) {
        // prefetch weight rows a few K-steps ahead (global_prefetch_b8)
        __builtin_prefetch(Wrow + (size_t)(lane & 15) * Kd + kk + 128, 0, 3);

        v16bf a[WMT];
        v16bf b[WNT];
#pragma unroll
        for (int i = 0; i < WMT; ++i)
            a[i] = load_frag_A(Arow + (size_t)(i * 16) * Kd, Kd, kk, lane);
#pragma unroll
        for (int j = 0; j < WNT; ++j)
            b[j] = load_frag_B(Wrow + (size_t)(j * 16) * Kd, Kd, kk, lane);
#pragma unroll
        for (int i = 0; i < WMT; ++i)
#pragma unroll
            for (int j = 0; j < WNT; ++j)
                acc[i][j] = __builtin_amdgcn_wmma_f32_16x16x32_bf16(
                    false, a[i], false, b[j], (short)0, acc[i][j], false, false);
    }

#pragma unroll
    for (int j = 0; j < WNT; ++j) {
        int   n  = nt * (16 * WNT) + j * 16 + (lane & 15);
        float bv = bias[n];
#pragma unroll
        for (int i = 0; i < WMT; ++i) {
            if (mode == 0) {
                float* O = (float*)out;
#pragma unroll
                for (int r = 0; r < 8; ++r) {
                    int m = mt * (16 * WMT) + i * 16 + r + ((lane >> 4) << 3);
                    O[(size_t)m * Nout + n] = acc[i][j][r] + bv;
                }
            } else {
                u16* O = (u16*)out;
                int h = n >> 6, d = n & 63;
#pragma unroll
                for (int r = 0; r < 8; ++r) {
                    int m  = mt * (16 * WMT) + i * 16 + r + ((lane >> 4) << 3);
                    int b  = m >> 11;           // m / NN  (NN = 2048)
                    int ni = m & 2047;          // m % NN
                    O[((((size_t)b * HH + h) * NN) + ni) * HD + d] = f2bf(acc[i][j][r] + bv);
                }
            }
        }
    }
}

// One wave per 16-query tile: full-N scores via WMMA (4 key tiles per pass,
// Q fragments reused across all of them) + per-query top-32.
__global__ void scores_topk_kernel(const u16* __restrict__ qb, const u16* __restrict__ kb,
                                   int* __restrict__ topk)
{
    __shared__ float stile[16][68];          // 16 queries x 64 keys (+pad)
    __shared__ float sl[16][TOPK];
    __shared__ int   il[16][TOPK];

    int bh   = blockIdx.y;              // b*H + h
    int qt   = blockIdx.x;              // query tile
    int lane = threadIdx.x;             // 32 threads (one wave)

    const u16* qbase = qb + (((size_t)bh * NN) + (size_t)qt * 16) * HD;
    const u16* kbase = kb + ((size_t)bh * NN) * HD;

    v16bf a0 = load_frag_A(qbase, HD, 0,  lane);
    v16bf a1 = load_frag_A(qbase, HD, 32, lane);

    int   filled = 0;
    float minv   = 0.0f;
    int   minp   = 0;

    for (int kt4 = 0; kt4 < NN / 64; ++kt4) {
        v8f acc[4] = {};
#pragma unroll
        for (int j = 0; j < 4; ++j) {
            const u16* ktile = kbase + (size_t)(kt4 * 64 + j * 16) * HD;
            v16bf b0 = load_frag_B(ktile, HD, 0,  lane);
            v16bf b1 = load_frag_B(ktile, HD, 32, lane);
            acc[j] = __builtin_amdgcn_wmma_f32_16x16x32_bf16(false, a0, false, b0,
                                                             (short)0, acc[j], false, false);
            acc[j] = __builtin_amdgcn_wmma_f32_16x16x32_bf16(false, a1, false, b1,
                                                             (short)0, acc[j], false, false);
        }
        int n = lane & 15;
#pragma unroll
        for (int j = 0; j < 4; ++j)
#pragma unroll
            for (int r = 0; r < 8; ++r)
                stile[r + ((lane >> 4) << 3)][j * 16 + n] = acc[j][r] * SCALE;
        __syncthreads();

        if (lane < 16) {
            for (int j = 0; j < 64; ++j) {
                float s  = stile[lane][j];
                int   ki = kt4 * 64 + j;
                if (filled < TOPK) {
                    sl[lane][filled] = s; il[lane][filled] = ki; ++filled;
                    if (filled == TOPK) {
                        minv = sl[lane][0]; minp = 0;
                        for (int t = 1; t < TOPK; ++t)
                            if (sl[lane][t] < minv) { minv = sl[lane][t]; minp = t; }
                    }
                } else if (s > minv) {
                    sl[lane][minp] = s; il[lane][minp] = ki;
                    minv = sl[lane][0]; minp = 0;
                    for (int t = 1; t < TOPK; ++t)
                        if (sl[lane][t] < minv) { minv = sl[lane][t]; minp = t; }
                }
            }
        }
        __syncthreads();
    }

    if (lane < 16) {
        int q = qt * 16 + lane;
        int* dst = topk + (((size_t)bh * NN) + q) * TOPK;
        for (int t = 0; t < TOPK; ++t) dst[t] = il[lane][t];
    }
}

// One wave per query: async-gather 32 key rows + 32 value rows into LDS
// (global_load_async_to_lds_b128, per-lane scattered addresses, ASYNCcnt),
// then softmax + weighted sum entirely out of LDS.
__global__ void local_attn_kernel(const u16* __restrict__ qb, const u16* __restrict__ kb,
                                  const u16* __restrict__ vb, const int* __restrict__ topk,
                                  u16* __restrict__ attn_out)
{
    __shared__ float wsm[8][TOPK];
    __shared__ u16   kstage[8][TOPK][HD];   // 32 KB
    __shared__ u16   vstage[8][TOPK][HD];   // 32 KB

    int wave = threadIdx.x >> 5;
    int lane = threadIdx.x & 31;
    size_t qg = (size_t)blockIdx.x * 8 + wave;   // 0 .. B*H*N-1
    int bh = (int)(qg / NN);
    int n  = (int)(qg % NN);
    int b  = bh / HH;
    int h  = bh % HH;

    const u16* qrow = qb + (((size_t)bh * NN) + n) * HD;
    int ki = topk[(((size_t)bh * NN) + n) * TOPK + lane];
    const u16* krow = kb + (((size_t)bh * NN) + ki) * HD;
    const u16* vrow = vb + (((size_t)bh * NN) + ki) * HD;

    // async gather: each lane owns one gathered key/value row (128 B each).
    // VDST holds the LDS byte offset (low 32 bits of the flat shared pointer).
#pragma unroll
    for (int c = 0; c < HD / 8; ++c) {
        unsigned kd = (unsigned)(uintptr_t)&kstage[wave][lane][c * 8];
        unsigned vd = (unsigned)(uintptr_t)&vstage[wave][lane][c * 8];
        unsigned long long ks = (unsigned long long)(uintptr_t)(krow + c * 8);
        unsigned long long vs = (unsigned long long)(uintptr_t)(vrow + c * 8);
        asm volatile("global_load_async_to_lds_b128 %0, %1, off"
                     :: "v"(kd), "v"(ks) : "memory");
        asm volatile("global_load_async_to_lds_b128 %0, %1, off"
                     :: "v"(vd), "v"(vs) : "memory");
    }
    asm volatile("s_wait_asynccnt 0" ::: "memory");

    float s = 0.0f;
    for (int d = 0; d < HD; ++d) s += bf2f(qrow[d]) * bf2f(kstage[wave][lane][d]);
    s *= SCALE;

    float m = s;
    for (int off = 16; off; off >>= 1) m = fmaxf(m, __shfl_xor(m, off, 32));
    float e = __expf(s - m);
    float t = e;
    for (int off = 16; off; off >>= 1) t += __shfl_xor(t, off, 32);
    float w = e / t;

    wsm[wave][lane] = w;
    __syncthreads();

    float o0 = 0.0f, o1 = 0.0f;
    for (int j = 0; j < TOPK; ++j) {
        float ww = wsm[wave][j];
        o0 += ww * bf2f(vstage[wave][j][lane]);
        o1 += ww * bf2f(vstage[wave][j][lane + 32]);
    }
    // layout [B, N, H*HD]
    size_t obase = (((size_t)b * NN) + n) * (HH * HD) + (size_t)h * HD;
    attn_out[obase + lane]      = f2bf(o0);
    attn_out[obase + lane + 32] = f2bf(o1);
}

extern "C" void kernel_launch(void* const* d_in, const int* in_sizes, int n_in,
                              void* d_out, int out_size, void* d_ws, size_t ws_size,
                              hipStream_t stream)
{
    const float* x  = (const float*)d_in[0];
    const float* Wq = (const float*)d_in[1];
    const float* bq = (const float*)d_in[2];
    const float* Wk = (const float*)d_in[3];
    const float* bk = (const float*)d_in[4];
    const float* Wv = (const float*)d_in[5];
    const float* bv = (const float*)d_in[6];
    const float* Wo = (const float*)d_in[7];
    const float* bo = (const float*)d_in[8];

    char*  ws  = (char*)d_ws;
    size_t off = 0;
    auto take = [&](size_t bytes) -> char* {
        char* p = ws + off;
        off += (bytes + 255) & ~(size_t)255;
        return p;
    };

    u16* xbf   = (u16*)take((size_t)MM * DIM * 2);
    u16* Wqb   = (u16*)take((size_t)DIM * DIM * 2);
    u16* Wkb   = (u16*)take((size_t)DIM * DIM * 2);
    u16* Wvb   = (u16*)take((size_t)DIM * DIM * 2);
    u16* Wob   = (u16*)take((size_t)DIM * DIM * 2);
    u16* qbuf  = (u16*)take((size_t)BB * HH * NN * HD * 2);
    u16* kbuf  = (u16*)take((size_t)BB * HH * NN * HD * 2);
    u16* vbuf  = (u16*)take((size_t)BB * HH * NN * HD * 2);
    int* topk  = (int*)take((size_t)BB * HH * NN * TOPK * 4);
    u16* attnb = (u16*)take((size_t)MM * DIM * 2);

    // 1) downconvert activations + weights to bf16
    {
        int n1 = MM * DIM;
        f32_to_bf16_kernel<<<(n1 + 255) / 256, 256, 0, stream>>>(x, xbf, n1);
        int n2 = DIM * DIM;
        f32_to_bf16_kernel<<<(n2 + 255) / 256, 256, 0, stream>>>(Wq, Wqb, n2);
        f32_to_bf16_kernel<<<(n2 + 255) / 256, 256, 0, stream>>>(Wk, Wkb, n2);
        f32_to_bf16_kernel<<<(n2 + 255) / 256, 256, 0, stream>>>(Wv, Wvb, n2);
        f32_to_bf16_kernel<<<(n2 + 255) / 256, 256, 0, stream>>>(Wo, Wob, n2);
    }

    // 2) Q/K/V projections: register-blocked WMMA GEMM, split-head bf16 out
    {
        int waves  = (MM / (16 * WMT)) * (DIM / (16 * WNT));   // 128*8 = 1024
        int blocks = waves / 4;                                // 4 waves / block
        wmma_gemm_kernel<<<blocks, 128, 0, stream>>>(xbf, Wqb, bq, qbuf, MM, DIM, DIM, 1);
        wmma_gemm_kernel<<<blocks, 128, 0, stream>>>(xbf, Wkb, bk, kbuf, MM, DIM, DIM, 1);
        wmma_gemm_kernel<<<blocks, 128, 0, stream>>>(xbf, Wvb, bv, vbuf, MM, DIM, DIM, 1);
    }

    // 3) dense scores (WMMA, 4-tile key blocking) + top-32 selection
    {
        dim3 grid(NN / 16, BB * HH);             // (128, 16)
        scores_topk_kernel<<<grid, 32, 0, stream>>>(qbuf, kbuf, topk);
    }

    // 4) gathered local softmax attention (async global->LDS gather)
    {
        int waves  = BB * HH * NN;               // 32768 queries
        local_attn_kernel<<<waves / 8, 256, 0, stream>>>(qbuf, kbuf, vbuf, topk, attnb);
    }

    // 5) output projection: WMMA GEMM -> fp32 d_out
    {
        int waves  = (MM / (16 * WMT)) * (DIM / (16 * WNT));
        int blocks = waves / 4;
        wmma_gemm_kernel<<<blocks, 128, 0, stream>>>(attnb, Wob, bo, d_out, MM, DIM, DIM, 0);
    }
}
